// MultiScaleDeformableAttentionBase_82884278879237
// MI455X (gfx1250) — matrix operand order repaired and checked by reading, not compiled, not run on hardware
//
#include <hip/hip_runtime.h>
#include <hip/hip_bf16.h>

// Multi-scale deformable attention for MI455X (gfx1250, wave32).
//
// Pipeline:
//   1) proj_gemm_wmma  : off    = query @ W_off^T  + b_off   (M=20000, N=256, K=256)
//   2) proj_gemm_wmma  : logits = query @ W_attn^T + b_attn  (M=20000, N=128, K=256)
//      -> both via v_wmma_f32_16x16x32_f16 (f16 inputs, f32 accumulate)
//   3) msda_sample     : per-(b,q,h) wave: shuffle softmax over 16 logits,
//      16 bilinear samples, lane==channel (HD=32 == wave32) so every corner
//      read is one coalesced 128B transaction; value (27MB) is L2-resident.

typedef __attribute__((ext_vector_type(16))) _Float16 v16h;
typedef __attribute__((ext_vector_type(8)))  float    v8f;
typedef __attribute__((ext_vector_type(4)))  float    f32x4;

#define BSc   2
#define NQc   10000
#define Dc    256
#define NHc   8
#define NLc   4
#define NPc   4
#define HDc   32
#define NVc   13294
#define Mrows (BSc * NQc)          // 20000, exactly 1250 16-row tiles

// ---------------------------------------------------------------------------
// Projection GEMM: C[M,N] = f16(A[M,K=256]) @ f16(W[N,K])^T + bias[N]
// One wave computes a 16x64 strip (4 accumulator tiles reusing the A fragment).
// ---------------------------------------------------------------------------
__global__ __launch_bounds__(32) void proj_gemm_wmma(
    const float* __restrict__ A,     // [M, 256]
    const float* __restrict__ Wt,    // [N, 256]
    const float* __restrict__ bias,  // [N]
    float* __restrict__ C,           // [M, N]
    int N)
{
    const int lane   = threadIdx.x;
    const int l15    = lane & 15;
    const int half16 = lane >> 4;                // 0: lanes 0-15, 1: lanes 16-31
    const int m0     = blockIdx.x * 16;
    const int n0     = blockIdx.y * 64;

    v8f acc[4] = {};

    const float* arow = A + (size_t)(m0 + l15) * Dc;

    for (int k0 = 0; k0 < Dc; k0 += 32) {
        // ---- A fragment: 16-bit A 16x32 layout (ISA 7.12.2) ----
        // lanes 0-15 : row=lane,    halves 0-7 = K[k0..k0+7],   8-15 = K[k0+16..k0+23]
        // lanes 16-31: row=lane-16, halves 0-7 = K[k0+8..+15],  8-15 = K[k0+24..+31]
        const int ca = k0 + (half16 << 3);
        f32x4 a0 = *(const f32x4*)(arow + ca);
        f32x4 a1 = *(const f32x4*)(arow + ca + 4);
        f32x4 a2 = *(const f32x4*)(arow + ca + 16);
        f32x4 a3 = *(const f32x4*)(arow + ca + 20);
        v16h afrag;
#pragma unroll
        for (int j = 0; j < 4; ++j) {
            afrag[j]      = (_Float16)a0[j];
            afrag[4 + j]  = (_Float16)a1[j];
            afrag[8 + j]  = (_Float16)a2[j];
            afrag[12 + j] = (_Float16)a3[j];
        }

        // ---- B fragments: 16-bit B 32x16, lane = N, same K order as A fragment ----
        const int cb = k0 + (half16 << 3);
#pragma unroll
        for (int nt = 0; nt < 4; ++nt) {
            const float* wrow = Wt + (size_t)(n0 + nt * 16 + l15) * Dc + cb;
            f32x4 b0 = *(const f32x4*)(wrow);
            f32x4 b1 = *(const f32x4*)(wrow + 4);
            f32x4 b2 = *(const f32x4*)(wrow + 16);
            f32x4 b3 = *(const f32x4*)(wrow + 20);
            v16h bfrag;
#pragma unroll
            for (int j = 0; j < 4; ++j) {
                bfrag[j]      = (_Float16)b0[j];
                bfrag[4 + j]  = (_Float16)b1[j];
                bfrag[8 + j]  = (_Float16)b2[j];
                bfrag[12 + j] = (_Float16)b3[j];
            }
            acc[nt] = __builtin_amdgcn_wmma_f32_16x16x32_f16(
                false, afrag, false, bfrag, (short)0, acc[nt], false, false);
        }
    }

    // ---- store: f32 C/D 16x16 layout: VGPR r -> M = r + 8*half16, N = lane&15 ----
#pragma unroll
    for (int nt = 0; nt < 4; ++nt) {
        const int col = n0 + nt * 16 + l15;
        const float bv = bias[col];
#pragma unroll
        for (int r = 0; r < 8; ++r) {
            const int row = m0 + r + half16 * 8;
            C[(size_t)row * N + col] = acc[nt][r] + bv;
        }
    }
}

// ---------------------------------------------------------------------------
// Fused softmax + bilinear sampling + weighted aggregation.
// One wave per (b, q, h); lane = channel (HD == 32 == wave32).
// ---------------------------------------------------------------------------
__global__ __launch_bounds__(128) void msda_sample(
    const float* __restrict__ off,     // [20000, 256]  (NH, NL*NP*2)
    const float* __restrict__ logits,  // [20000, 128]  (NH, NL*NP)
    const float* __restrict__ refpt,   // [BS, NQ, NL, 2]
    const float* __restrict__ value,   // [BS, NV, NH, HD]
    float* __restrict__ out)           // [BS, NH*HD, NQ]
{
    const int wid  = blockIdx.x * 4 + (threadIdx.x >> 5);
    const int lane = threadIdx.x & 31;
    const int h    = wid % NHc;
    const int q    = (wid / NHc) % NQc;
    const int b    = wid / (NHc * NQc);
    const int bq   = b * NQc + q;

    // level constants (compile-time; folded by the unrolled sample loop)
    const int   kW[NLc]  = {100, 50, 25, 13};
    const int   kH[NLc]  = {100, 50, 25, 13};
    const int   kSt[NLc] = {0, 10000, 12500, 13125};

    // 32 offsets per (b,q,h): index = l*8 + p*2 + xy  -> one per lane
    const float offv = off[(size_t)bq * 256 + h * 32 + lane];

    // softmax over the 16 (l,p) logits, replicated in both half-waves
    float lg = logits[(size_t)bq * 128 + h * 16 + (lane & 15)];
    float mx = lg;
#pragma unroll
    for (int d = 8; d >= 1; d >>= 1) mx = fmaxf(mx, __shfl_xor(mx, d, 16));
    const float e = __expf(lg - mx);
    float ssum = e;
#pragma unroll
    for (int d = 8; d >= 1; d >>= 1) ssum += __shfl_xor(ssum, d, 16);
    const float aw = e / ssum;

    // lane (j = lane&15) computes image-space position of sample j.
    // loc = (ref + off/W)*2-1 then x=(loc+1)*W/2-0.5  ==  ref*W + off - 0.5
    const int   s    = lane & 15;
    const int   lv   = s >> 2;
    const float ox   = __shfl(offv, 2 * s, 32);
    const float oy   = __shfl(offv, 2 * s + 1, 32);
    const float rx   = refpt[((size_t)bq * NLc + lv) * 2 + 0];
    const float ry   = refpt[((size_t)bq * NLc + lv) * 2 + 1];
    const float xi   = rx * (float)kW[lv] + ox - 0.5f;
    const float yi   = ry * (float)kH[lv] + oy - 0.5f;

    // accumulate over the 16 samples; this lane owns channel c = lane
    const float* vbase =
        value + ((size_t)b * NVc * NHc + (size_t)h) * HDc + lane;
    float acc = 0.0f;
#pragma unroll
    for (int j = 0; j < 16; ++j) {
        const float xs = __shfl(xi, j, 32);
        const float ys = __shfl(yi, j, 32);
        const float wj = __shfl(aw, j, 32);
        const int   lvl = j >> 2;
        const int   Wi = kW[lvl], Hi = kH[lvl];
        const float* vb = vbase + (size_t)kSt[lvl] * (NHc * HDc);

        const float xf = floorf(xs), yf = floorf(ys);
        const int   x0 = (int)xf,    y0 = (int)yf;
        const float fx = xs - xf,    fy = ys - yf;
        const float w00 = (1.0f - fx) * (1.0f - fy);
        const float w10 = fx * (1.0f - fy);
        const float w01 = (1.0f - fx) * fy;
        const float w11 = fx * fy;

        const bool xin0 = (x0 >= 0) && (x0 < Wi);
        const bool xin1 = (x0 + 1 >= 0) && (x0 + 1 < Wi);
        const bool yin0 = (y0 >= 0) && (y0 < Hi);
        const bool yin1 = (y0 + 1 >= 0) && (y0 + 1 < Hi);

        float v00 = 0.0f, v10 = 0.0f, v01 = 0.0f, v11 = 0.0f;
        if (xin0 && yin0) v00 = vb[(size_t)(y0 * Wi + x0) * (NHc * HDc)];
        if (xin1 && yin0) v10 = vb[(size_t)(y0 * Wi + x0 + 1) * (NHc * HDc)];
        if (xin0 && yin1) v01 = vb[(size_t)((y0 + 1) * Wi + x0) * (NHc * HDc)];
        if (xin1 && yin1) v11 = vb[(size_t)((y0 + 1) * Wi + x0 + 1) * (NHc * HDc)];

        acc += wj * (w00 * v00 + w10 * v10 + w01 * v01 + w11 * v11);
    }

    // out[b, h*HD + c, q]
    out[((size_t)b * (NHc * HDc) + h * HDc + lane) * NQc + q] = acc;
}

extern "C" void kernel_launch(void* const* d_in, const int* in_sizes, int n_in,
                              void* d_out, int out_size, void* d_ws, size_t ws_size,
                              hipStream_t stream) {
    (void)in_sizes; (void)n_in; (void)out_size; (void)ws_size;
    const float* query  = (const float*)d_in[0];
    const float* value  = (const float*)d_in[1];
    const float* refpt  = (const float*)d_in[2];
    // d_in[3] spatial_shapes: compile-time constants (FEATS fixed)
    const float* W_off  = (const float*)d_in[4];
    const float* b_off  = (const float*)d_in[5];
    const float* W_attn = (const float*)d_in[6];
    const float* b_attn = (const float*)d_in[7];
    float* out = (float*)d_out;

    // workspace: off [20000*256] f32, then logits [20000*128] f32 (~31 MB)
    float* ws_off  = (float*)d_ws;
    float* ws_attn = ws_off + (size_t)Mrows * 256;

    dim3 blk(32);
    dim3 g_off(Mrows / 16, 256 / 64);    // 1250 x 4
    proj_gemm_wmma<<<g_off, blk, 0, stream>>>(query, W_off, b_off, ws_off, 256);
    dim3 g_att(Mrows / 16, 128 / 64);    // 1250 x 2
    proj_gemm_wmma<<<g_att, blk, 0, stream>>>(query, W_attn, b_attn, ws_attn, 128);

    const int waves = BSc * NQc * NHc;   // 160000, 4 waves per block
    msda_sample<<<waves / 4, 128, 0, stream>>>(ws_off, ws_attn, refpt, value, out);
}
